// SNN_18433999634702
// MI455X (gfx1250) — matrix operand order repaired and checked by reading, not compiled, not run on hardware
//
#include <hip/hip_runtime.h>

// CDNA5 / gfx1250: wave32, WMMA bf16 path.
typedef __attribute__((ext_vector_type(16))) __bf16 v16bf;
typedef __attribute__((ext_vector_type(8)))  float  v8f;

constexpr int T = 64, B = 2048, F = 784, O = 10;
constexpr int NPAD   = 16;               // pad O=10 -> 16 (WMMA N)
constexpr int ROWS   = T * B;            // 131072 flattened GEMM rows
constexpr int TILES  = ROWS / 16;        // 8192 16-row tiles (one per wave)
constexpr int CHUNKS = (F + 31) / 32;    // 25 K-chunks of 32 (last half-valid)

union Frag { uint4 u[2]; v16bf v; };

// f32 -> bf16 with round-to-nearest-even (matches v_cvt rounding)
__device__ __forceinline__ unsigned short f2bf_rne(float f) {
  union { float f; unsigned u; } x; x.f = f;
  unsigned u = x.u;
  u += 0x7FFFu + ((u >> 16) & 1u);
  return (unsigned short)(u >> 16);
}

// ---------------------------------------------------------------------------
// Pass 1: C[row][0..15] = bf16-WMMA( X[row][0..783] , W[n][k] ), row = t*B+b.
// One wave per 16-row tile. W staged in LDS in exact B-fragment layout.
// ---------------------------------------------------------------------------
__global__ __launch_bounds__(128) void snn_gemm_bf16_wmma(
    const float* __restrict__ X, const float* __restrict__ W,
    float* __restrict__ C) {
  // B fragments: CHUNKS chunks x 32 lanes x 8 dwords (16 bf16) = 25 KB
  __shared__ __align__(16) unsigned ldsB[CHUNKS * 32 * 8];

  // Stage W (O x F, f32) -> bf16 B-matrix fragments, zero-padded (n>=O, k>=F).
  // Layout per CDNA5 ISA: lanes 0-15 hold K 0-7,16-23 of the chunk
  // (2 packed per dword), lanes 16-31 hold K 8-15,24-31; N = lane & 15.
  for (int idx = threadIdx.x; idx < CHUNKS * 32 * 8; idx += blockDim.x) {
    int c    = idx >> 8;          // chunk
    int rem  = idx & 255;
    int lane = rem >> 3;
    int vreg = rem & 7;
    int n = lane & 15;
    int k = c * 32 + ((lane >> 4) << 3) + ((vreg >> 2) << 4) + 2 * (vreg & 3);
    float f0 = (n < O && k     < F) ? W[n * F + k]     : 0.0f;
    float f1 = (n < O && (k+1) < F) ? W[n * F + k + 1] : 0.0f;
    ldsB[idx] = (unsigned)f2bf_rne(f0) | ((unsigned)f2bf_rne(f1) << 16);
  }
  __syncthreads();

  const int lane = threadIdx.x & 31;
  const int tile = blockIdx.x * (blockDim.x >> 5) + (threadIdx.x >> 5);
  const int hi   = lane >> 4;      // 0: K 0-7/16-23 ; 1: K 8-15/24-31
  const int kgrp = hi * 8;

  const float* rowp = X + (size_t)(tile * 16 + (lane & 15)) * F;

  v8f acc = {};

#pragma unroll 4
  for (int c = 0; c < CHUNKS - 1; ++c) {
    // A fragment: 16 f32 per lane (4 x global_load_b128), cvt -> bf16
    const float4 a0 = *(const float4*)(rowp + c * 32 + kgrp);
    const float4 a1 = *(const float4*)(rowp + c * 32 + kgrp + 4);
    const float4 a2 = *(const float4*)(rowp + c * 32 + 16 + kgrp);
    const float4 a3 = *(const float4*)(rowp + c * 32 + 16 + kgrp + 4);
    Frag a;
    a.v[0]=(__bf16)a0.x; a.v[1]=(__bf16)a0.y; a.v[2]=(__bf16)a0.z; a.v[3]=(__bf16)a0.w;
    a.v[4]=(__bf16)a1.x; a.v[5]=(__bf16)a1.y; a.v[6]=(__bf16)a1.z; a.v[7]=(__bf16)a1.w;
    a.v[8]=(__bf16)a2.x; a.v[9]=(__bf16)a2.y; a.v[10]=(__bf16)a2.z; a.v[11]=(__bf16)a2.w;
    a.v[12]=(__bf16)a3.x; a.v[13]=(__bf16)a3.y; a.v[14]=(__bf16)a3.z; a.v[15]=(__bf16)a3.w;

    Frag bm;
    const uint4* bp = (const uint4*)&ldsB[c * 256 + lane * 8];  // 2 x ds_load_b128
    bm.u[0] = bp[0]; bm.u[1] = bp[1];

    acc = __builtin_amdgcn_wmma_f32_16x16x32_bf16(
        false, a.v, false, bm.v, (short)0, acc, false, false);
  }

  // Epilogue chunk 24: K 768..783 valid, 784..799 zero (A elems 8..15 = 0;
  // matching LDS B entries already zero-filled).
  {
    const int c = CHUNKS - 1;
    const float4 a0 = *(const float4*)(rowp + c * 32 + kgrp);
    const float4 a1 = *(const float4*)(rowp + c * 32 + kgrp + 4);
    Frag a;
    a.v[0]=(__bf16)a0.x; a.v[1]=(__bf16)a0.y; a.v[2]=(__bf16)a0.z; a.v[3]=(__bf16)a0.w;
    a.v[4]=(__bf16)a1.x; a.v[5]=(__bf16)a1.y; a.v[6]=(__bf16)a1.z; a.v[7]=(__bf16)a1.w;
#pragma unroll
    for (int i = 8; i < 16; ++i) a.v[i] = (__bf16)0.0f;

    Frag bm;
    const uint4* bp = (const uint4*)&ldsB[c * 256 + lane * 8];
    bm.u[0] = bp[0]; bm.u[1] = bp[1];

    acc = __builtin_amdgcn_wmma_f32_16x16x32_bf16(
        false, a.v, false, bm.v, (short)0, acc, false, false);
  }

  // D layout: VGPR r holds M = r + 8*hi, N = lane & 15 -> coalesced 64B rows.
  const int mbase = tile * 16 + hi * 8;
  const int ncol  = lane & 15;
  float* cp = C + (size_t)mbase * NPAD + ncol;
#pragma unroll
  for (int r = 0; r < 8; ++r) cp[r * NPAD] = acc[r];
}

// ---------------------------------------------------------------------------
// Pass 2: per-(b,o) IF-neuron scan over T, emit firing rate.
// ---------------------------------------------------------------------------
__global__ __launch_bounds__(128) void snn_scan(
    const float* __restrict__ C, const float* __restrict__ bias,
    float* __restrict__ out) {
  int idx = blockIdx.x * blockDim.x + threadIdx.x;
  if (idx >= B * O) return;
  int b = idx / O;
  int o = idx - b * O;
  float bo = bias[o];
  const float* p = C + (size_t)b * NPAD + o;
  float v = 0.0f, acc = 0.0f;
#pragma unroll 8
  for (int t = 0; t < T; ++t) {
    v += p[(size_t)t * B * NPAD] + bo;   // charge
    if (v >= 1.0f) { acc += 1.0f; v = 0.0f; }  // fire + hard reset
  }
  out[idx] = acc * (1.0f / (float)T);
}

extern "C" void kernel_launch(void* const* d_in, const int* in_sizes, int n_in,
                              void* d_out, int out_size, void* d_ws, size_t ws_size,
                              hipStream_t stream) {
  (void)in_sizes; (void)n_in; (void)out_size; (void)ws_size;
  const float* X    = (const float*)d_in[0];  // [T,B,F] f32
  const float* W    = (const float*)d_in[1];  // [O,F]   f32
  const float* bias = (const float*)d_in[2];  // [O]     f32
  float* C   = (float*)d_ws;                  // [T*B, 16] f32 = 8 MiB scratch
  float* out = (float*)d_out;                 // [B,O] f32

  dim3 blk1(128), grd1(TILES / 4);            // 4 waves/WG, 1 tile/wave
  hipLaunchKernelGGL(snn_gemm_bf16_wmma, grd1, blk1, 0, stream, X, W, C);

  dim3 blk2(128), grd2((B * O + 127) / 128);
  hipLaunchKernelGGL(snn_scan, grd2, blk2, 0, stream, C, bias, out);
}
